// CausalAttNet_70592082477597
// MI455X (gfx1250) — matrix operand order repaired and checked by reading, not compiled, not run on hardware
//
#include <hip/hip_runtime.h>
#include <hip/hip_bf16.h>

// ---------------------------------------------------------------------------
// Problem constants (from the reference)
// ---------------------------------------------------------------------------
#define NG      1024          // groups
#define NPG     64            // nodes per group
#define EPG     256           // edges per group
#define NN      (NG * NPG)    // 65536 nodes
#define NE      (NG * EPG)    // 262144 edges
#define INCH    768
#define CH      128
#define KSEL    128           // top-K per group (RATIO=0.5)
#define MLP_IN  (2 * CH)      // 256
#define MLP_HID (4 * CH)      // 512

typedef __attribute__((ext_vector_type(16))) __bf16 v16bf;
typedef __attribute__((ext_vector_type(8)))  float  v8f;

__device__ __forceinline__ float relu_f(float v) { return v > 0.f ? v : 0.f; }

union pack4bf {
    __bf16 h[4];
    unsigned long long u;
};

// ---------------------------------------------------------------------------
// Weight swizzle: fp32 (K x (CA+CB)) -> bf16 fragments laid out so that a
// wave32 B-fragment (V_WMMA_F32_16X16X32_BF16) is 32 contiguous bytes/lane:
//   out[((kt*NT + nt)*32 + lane)*16 + j]  ==  W[kt*32 + (lane>>4)*16 + j][nt*16 + (lane&15)]
// ---------------------------------------------------------------------------
__global__ void swizzle_weights(const float* __restrict__ W, const float* __restrict__ V,
                                int K, int CA, int CB, __bf16* __restrict__ out) {
    int NT = (CA + CB) >> 4;
    int total = (K >> 5) * NT * 32;
    int idx = blockIdx.x * blockDim.x + threadIdx.x;
    if (idx >= total) return;
    int lane = idx & 31;
    int tile = idx >> 5;
    int nt = tile % NT;
    int kt = tile / NT;
    int kbase = kt * 32 + ((lane >> 4) << 4);
    int n = nt * 16 + (lane & 15);
    __bf16* dst = out + (size_t)idx * 16;
    if (n < CA) {
        const float* src = W + (size_t)kbase * CA + n;
        #pragma unroll
        for (int j = 0; j < 16; ++j) dst[j] = (__bf16)src[(size_t)j * CA];
    } else {
        int nc = n - CA;
        const float* src = V + (size_t)kbase * CB + nc;
        #pragma unroll
        for (int j = 0; j < 16; ++j) dst[j] = (__bf16)src[(size_t)j * CB];
    }
}

// ---------------------------------------------------------------------------
// deg[col] += ew ;  norm[e] = rsqrt(deg[row]) * ew * rsqrt(deg[col])
// ---------------------------------------------------------------------------
__global__ void deg_kernel(const int* __restrict__ ei, const float* __restrict__ ew,
                           float* __restrict__ deg) {
    int e = blockIdx.x * blockDim.x + threadIdx.x;
    if (e >= NE) return;
    atomicAdd(&deg[ei[NE + e]], ew[e]);
}

__global__ void norm_kernel(const int* __restrict__ ei, const float* __restrict__ ew,
                            const float* __restrict__ deg, float* __restrict__ nrm) {
    int e = blockIdx.x * blockDim.x + threadIdx.x;
    if (e >= NE) return;
    float dr = deg[ei[e]];
    float dc = deg[ei[NE + e]];
    float ir = dr > 0.f ? rsqrtf(dr) : 0.f;
    float ic = dc > 0.f ? rsqrtf(dc) : 0.f;
    nrm[e] = ir * ew[e] * ic;
}

// ---------------------------------------------------------------------------
// Tiled WMMA GEMM:  Out(16 rows x NT*16 cols, f32) = X(16 x K) * Wsw
// Block = 128 threads = 4 waves; each wave owns 4 column tiles.
// A is double-buffered in LDS in A-fragment order; all 4 B fragments of a
// K-step are preloaded before the WMMA chain (single wait, 4 back-to-back
// v_wmma). Next K-step's A gather is issued before the WMMAs.
// ---------------------------------------------------------------------------
template <typename TIn>
__global__ void gemm16_kernel(const TIn* __restrict__ X, const __bf16* __restrict__ Wsw,
                              float* __restrict__ Out, int K, int NT) {
    __shared__ __bf16 As[2][32 * 16];
    const int tid = threadIdx.x;
    const int wave = tid >> 5, lane = tid & 31;
    const int row0 = blockIdx.x * 16;
    const int ncols = NT * 16;
    const int ksteps = K >> 5;

    // per-thread A staging coordinates (fixed across K-steps)
    const int base = tid * 4;
    const int alane = base >> 4, aj = base & 15;
    const int am = alane & 15, ah = alane >> 4;
    const int KlBase = aj + 16 * ah;
    const TIn* srcRow = X + (size_t)(row0 + am) * K + KlBase;
    __bf16* myslot0 = &As[0][alane * 16 + aj];
    __bf16* myslot1 = &As[1][alane * 16 + aj];

    v8f acc[4];
    #pragma unroll
    for (int t = 0; t < 4; ++t) acc[t] = (v8f)0.f;

    // prologue: stage K-step 0
    {
        pack4bf pk;
        #pragma unroll
        for (int q = 0; q < 4; ++q) pk.h[q] = (__bf16)(float)srcRow[q];
        *reinterpret_cast<unsigned long long*>(myslot0) = pk.u;
    }
    __syncthreads();

    for (int ks = 0; ks < ksteps; ++ks) {
        const bool have_next = (ks + 1) < ksteps;
        // issue next K-step's A gather early (overlaps with WMMAs below)
        pack4bf nxt;
        if (have_next) {
            const TIn* s = srcRow + (size_t)(ks + 1) * 32;
            #pragma unroll
            for (int q = 0; q < 4; ++q) nxt.h[q] = (__bf16)(float)s[q];
            if ((ks + 2) < ksteps)
                __builtin_prefetch(srcRow + (size_t)(ks + 2) * 32, 0, 3);
        }

        v16bf a = *reinterpret_cast<const v16bf*>(&As[ks & 1][lane * 16]);
        // preload all 4 B fragments, then chain the WMMAs
        v16bf bfr[4];
        #pragma unroll
        for (int t = 0; t < 4; ++t) {
            int nt = wave * 4 + t;
            bfr[t] = *reinterpret_cast<const v16bf*>(
                &Wsw[(((size_t)ks * NT + nt) * 32 + lane) * 16]);
        }
        #pragma unroll
        for (int t = 0; t < 4; ++t)
            acc[t] = __builtin_amdgcn_wmma_f32_16x16x32_bf16(
                false, a, false, bfr[t], (short)0, acc[t], false, false);

        if (have_next) {
            __bf16* slot = ((ks + 1) & 1) ? myslot1 : myslot0;
            *reinterpret_cast<unsigned long long*>(slot) = nxt.u;
        }
        __syncthreads();
    }

    // store C: lane<16 -> rows v, lane>=16 -> rows v+8 ; col = tilebase + (lane&15)
    const int hh = lane >> 4, n0 = lane & 15;
    #pragma unroll
    for (int t = 0; t < 4; ++t) {
        int n = (wave * 4 + t) * 16 + n0;
        #pragma unroll
        for (int v = 0; v < 8; ++v)
            Out[(size_t)(row0 + v + 8 * hh) * ncols + n] = acc[t][v];
    }
}

// ---------------------------------------------------------------------------
// Scatter:  agg[col, c] += norm[e] * M[row, c]   (M has 256 cols, use 0..127)
// ---------------------------------------------------------------------------
__global__ void scatter_kernel(const float* __restrict__ M, const int* __restrict__ ei,
                               const float* __restrict__ nrm, float* __restrict__ agg) {
    int e = blockIdx.x;
    int c = threadIdx.x;
    int r = ei[e], col = ei[NE + e];
    float v = nrm[e] * M[(size_t)r * 256 + c];
    atomicAdd(&agg[(size_t)col * CH + c], v);
}

// ---------------------------------------------------------------------------
// Epilogues: h = relu(agg + hV + b); layer-1 stores bf16, layer-2 also writes
// the two x2 output copies (f32).
// ---------------------------------------------------------------------------
__global__ void epilogue1_kernel(const float* __restrict__ agg, const float* __restrict__ out1,
                                 const float* __restrict__ b, __bf16* __restrict__ hbf) {
    size_t i = (size_t)blockIdx.x * blockDim.x + threadIdx.x;
    if (i >= (size_t)NN * CH) return;
    size_t n = i >> 7; int c = (int)(i & 127);
    float v = relu_f(agg[i] + out1[n * 256 + 128 + c] + b[c]);
    hbf[i] = (__bf16)v;
}

__global__ void epilogue2_kernel(const float* __restrict__ agg, const float* __restrict__ out2,
                                 const float* __restrict__ b, __bf16* __restrict__ hbf,
                                 float* __restrict__ outA, float* __restrict__ outB) {
    size_t i = (size_t)blockIdx.x * blockDim.x + threadIdx.x;
    if (i >= (size_t)NN * CH) return;
    size_t n = i >> 7; int c = (int)(i & 127);
    float v = relu_f(agg[i] + out2[n * 256 + 128 + c] + b[c]);
    hbf[i] = (__bf16)v;
    outA[i] = v;
    outB[i] = v;
}

// ---------------------------------------------------------------------------
// Edge MLP: per block = 16 edges. A = concat(x2[row], x2[col]) (16x256 bf16),
// double-buffered in LDS; 8 B fragments preloaded per K-step; relu + bias and
// the 512->1 second layer fused into the epilogue via LDS float atomics.
// ---------------------------------------------------------------------------
__global__ void mlp_edge_kernel(const __bf16* __restrict__ x2bf, const int* __restrict__ ei,
                                const __bf16* __restrict__ mw1sw, const float* __restrict__ mb1,
                                const float* __restrict__ mw2, const float* __restrict__ mb2,
                                float* __restrict__ predWs, float* __restrict__ predOut) {
    __shared__ __bf16 As[2][32 * 16];
    __shared__ float sPred[16];
    const int tid = threadIdx.x;
    const int wave = tid >> 5, lane = tid & 31;
    const int et = blockIdx.x; // edge tile (16 edges)
    if (tid < 16) sPred[tid] = mb2[0];

    // per-thread A staging coordinates
    const int base = tid * 4;
    const int alane = base >> 4, aj = base & 15;
    const int am = alane & 15, ah = alane >> 4;
    const int KlBase = aj + 16 * ah;
    const int e_stage = et * 16 + am;
    const int nrow = ei[e_stage];
    const int ncol = ei[NE + e_stage];
    __bf16* myslot0 = &As[0][alane * 16 + aj];
    __bf16* myslot1 = &As[1][alane * 16 + aj];

    v8f acc[8];
    #pragma unroll
    for (int t = 0; t < 8; ++t) acc[t] = (v8f)0.f;

    // prologue: stage K-step 0 (Kl < 128 -> row side)
    {
        const __bf16* s = x2bf + (size_t)nrow * CH + KlBase;
        pack4bf pk;
        #pragma unroll
        for (int q = 0; q < 4; ++q) pk.h[q] = s[q];
        *reinterpret_cast<unsigned long long*>(myslot0) = pk.u;
    }
    __syncthreads();

    for (int ks = 0; ks < 8; ++ks) { // K = 256, 32 per step
        const bool have_next = (ks + 1) < 8;
        pack4bf nxt;
        if (have_next) {
            int Kl = KlBase + (ks + 1) * 32;
            int node = (Kl < CH) ? nrow : ncol;
            const __bf16* s = x2bf + (size_t)node * CH + (Kl & (CH - 1));
            #pragma unroll
            for (int q = 0; q < 4; ++q) nxt.h[q] = s[q];
        }

        v16bf a = *reinterpret_cast<const v16bf*>(&As[ks & 1][lane * 16]);
        v16bf bfr[8];
        #pragma unroll
        for (int t = 0; t < 8; ++t) {
            int nt = wave * 8 + t; // 32 column tiles total (512 cols)
            bfr[t] = *reinterpret_cast<const v16bf*>(
                &mw1sw[(((size_t)ks * 32 + nt) * 32 + lane) * 16]);
        }
        #pragma unroll
        for (int t = 0; t < 8; ++t)
            acc[t] = __builtin_amdgcn_wmma_f32_16x16x32_bf16(
                false, a, false, bfr[t], (short)0, acc[t], false, false);

        if (have_next) {
            __bf16* slot = ((ks + 1) & 1) ? myslot1 : myslot0;
            *reinterpret_cast<unsigned long long*>(slot) = nxt.u;
        }
        __syncthreads();
    }

    // Fused second layer: pred[row] += sum_n relu(h + mb1[n]) * mw2[n]
    const int hh = lane >> 4, n0 = lane & 15;
    float part[8];
    #pragma unroll
    for (int v = 0; v < 8; ++v) part[v] = 0.f;
    #pragma unroll
    for (int t = 0; t < 8; ++t) {
        int n = wave * 128 + t * 16 + n0;
        float wv = mw2[n];
        float bv = mb1[n];
        #pragma unroll
        for (int v = 0; v < 8; ++v) {
            float h = relu_f(acc[t][v] + bv);
            part[v] += h * wv;
        }
    }
    #pragma unroll
    for (int v = 0; v < 8; ++v) atomicAdd(&sPred[v + 8 * hh], part[v]);
    __syncthreads();
    if (tid < 16) {
        int e = et * 16 + tid;
        float p = sPred[tid];
        predWs[e] = p;
        predOut[e] = p;
    }
}

// ---------------------------------------------------------------------------
// batch copy (int -> f32, two output slots)
// ---------------------------------------------------------------------------
__global__ void batch_kernel(const int* __restrict__ batch, float* __restrict__ oA,
                             float* __restrict__ oB) {
    int n = blockIdx.x * blockDim.x + threadIdx.x;
    if (n >= NN) return;
    float v = (float)batch[n];
    oA[n] = v;
    oB[n] = v;
}

// ---------------------------------------------------------------------------
// Per-group descending stable sort (bitonic on 64-bit keys) + top-K split.
// key = (~monotone(pred) << 32) | local_idx  -> ascending sort == reference
// argsort(-pred) with stable tie-break on index.
// ---------------------------------------------------------------------------
__device__ __forceinline__ unsigned f2mono(float f) {
    unsigned u = __float_as_uint(f);
    return (u & 0x80000000u) ? ~u : (u | 0x80000000u);
}

__global__ void topk_kernel(const float* __restrict__ pred, const int* __restrict__ ei,
                            const float* __restrict__ ea, float* __restrict__ out,
                            size_t o_eic, size_t o_eac, size_t o_pwc,
                            size_t o_eid, size_t o_ead, size_t o_pwd) {
    __shared__ unsigned long long keys[EPG];
    const int g = blockIdx.x;
    const int t = threadIdx.x;
    {
        float p = pred[g * EPG + t];
        keys[t] = ((unsigned long long)(~f2mono(p)) << 32) | (unsigned)t;
    }
    __syncthreads();
    for (unsigned size = 2; size <= EPG; size <<= 1) {
        for (unsigned stride = size >> 1; stride > 0; stride >>= 1) {
            unsigned partner = (unsigned)t ^ stride;
            if (partner > (unsigned)t) {
                unsigned long long a = keys[t], b = keys[partner];
                bool up = ((t & size) == 0);
                if ((a > b) == up) { keys[t] = b; keys[partner] = a; }
            }
            __syncthreads();
        }
    }
    unsigned idx = (unsigned)keys[t]; // local edge index within group
    int eid = g * EPG + (int)idx;
    float p = pred[eid];
    float r0 = (float)ei[eid];
    float r1 = (float)ei[NE + eid];
    float a = ea[eid];
    const size_t GK = (size_t)NG * KSEL;
    if (t < KSEL) {
        size_t pos = (size_t)g * KSEL + t;
        out[o_eic + pos] = r0;
        out[o_eic + GK + pos] = r1;
        out[o_eac + pos] = a;
        out[o_pwc + pos] = p;
    } else {
        size_t pos = (size_t)g * KSEL + (t - KSEL);
        out[o_eid + pos] = r0;
        out[o_eid + GK + pos] = r1;
        out[o_ead + pos] = a;
        out[o_pwd + pos] = -p;
    }
}

// ---------------------------------------------------------------------------
// Host launch
// ---------------------------------------------------------------------------
extern "C" void kernel_launch(void* const* d_in, const int* in_sizes, int n_in,
                              void* d_out, int out_size, void* d_ws, size_t ws_size,
                              hipStream_t stream) {
    const float* x    = (const float*)d_in[0];
    const int*   ei   = (const int*)d_in[1];
    const float* ea   = (const float*)d_in[2];
    const int*   bat  = (const int*)d_in[3];
    const float* W1   = (const float*)d_in[4];
    const float* V1   = (const float*)d_in[5];
    const float* b1   = (const float*)d_in[6];
    const float* W2   = (const float*)d_in[7];
    const float* V2   = (const float*)d_in[8];
    const float* b2   = (const float*)d_in[9];
    const float* mw1  = (const float*)d_in[10];
    const float* mb1  = (const float*)d_in[11];
    const float* mw2  = (const float*)d_in[12];
    const float* mb2  = (const float*)d_in[13];
    float* out = (float*)d_out;
    char*  ws  = (char*)d_ws;

    // workspace layout (bytes)
    __bf16* w1v1sw = (__bf16*)(ws + 0);              // 768*256*2   = 393216
    __bf16* w2v2sw = (__bf16*)(ws + 393216);         // 128*256*2   = 65536
    __bf16* mw1sw  = (__bf16*)(ws + 458752);         // 256*512*2   = 262144
    float*  deg    = (float*) (ws + 720896);         // NN*4        = 262144
    float*  nrm    = (float*) (ws + 983040);         // NE*4        = 1048576
    float*  out1   = (float*) (ws + 2031616);        // NN*256*4    = 67108864 (reused as out2)
    float*  agg    = (float*) (ws + 69140480);       // NN*128*4    = 33554432 (reused)
    __bf16* hbf    = (__bf16*)(ws + 102694912);      // NN*128*2    = 16777216 (x1 then x2)
    float*  predWs = (float*) (ws + 119472128);      // NE*4        = 1048576

    // output element offsets
    const size_t X2SZ = (size_t)NN * CH;       // 8388608
    const size_t EI2  = 2 * (size_t)NG * KSEL; // 262144
    const size_t EA1  = (size_t)NG * KSEL;     // 131072
    const size_t o_x2a  = 0;
    const size_t o_eic  = o_x2a + X2SZ;
    const size_t o_eac  = o_eic + EI2;
    const size_t o_pwc  = o_eac + EA1;
    const size_t o_batc = o_pwc + EA1;
    const size_t o_x2b  = o_batc + NN;
    const size_t o_eid  = o_x2b + X2SZ;
    const size_t o_ead  = o_eid + EI2;
    const size_t o_pwd  = o_ead + EA1;
    const size_t o_batd = o_pwd + EA1;
    const size_t o_pred = o_batd + NN;

    // --- prep ---
    hipMemsetAsync(deg, 0, (size_t)NN * 4, stream);
    hipMemsetAsync(agg, 0, (size_t)NN * CH * 4, stream);
    swizzle_weights<<<(24 * 16 * 32 + 255) / 256, 256, 0, stream>>>(W1, V1, INCH, CH, CH, w1v1sw);
    swizzle_weights<<<(4 * 16 * 32 + 255) / 256, 256, 0, stream>>>(W2, V2, CH, CH, CH, w2v2sw);
    swizzle_weights<<<(8 * 32 * 32 + 255) / 256, 256, 0, stream>>>(mw1, nullptr, MLP_IN, MLP_HID, 0, mw1sw);
    deg_kernel<<<NE / 256, 256, 0, stream>>>(ei, ea, deg);
    norm_kernel<<<NE / 256, 256, 0, stream>>>(ei, ea, deg, nrm);

    // --- ARMA layer 1 ---
    gemm16_kernel<float><<<NN / 16, 128, 0, stream>>>(x, w1v1sw, out1, INCH, 16);
    scatter_kernel<<<NE, CH, 0, stream>>>(out1, ei, nrm, agg);
    epilogue1_kernel<<<(NN * CH) / 256, 256, 0, stream>>>(agg, out1, b1, hbf);

    // --- ARMA layer 2 ---
    hipMemsetAsync(agg, 0, (size_t)NN * CH * 4, stream);
    gemm16_kernel<__bf16><<<NN / 16, 128, 0, stream>>>(hbf, w2v2sw, out1, CH, 16);
    scatter_kernel<<<NE, CH, 0, stream>>>(out1, ei, nrm, agg);
    epilogue2_kernel<<<(NN * CH) / 256, 256, 0, stream>>>(agg, out1, b2, hbf,
                                                          out + o_x2a, out + o_x2b);

    // --- edge MLP ---
    mlp_edge_kernel<<<NE / 16, 128, 0, stream>>>(hbf, ei, mw1sw, mb1, mw2, mb2,
                                                 predWs, out + o_pred);

    // --- batch copies + per-group top-K split ---
    batch_kernel<<<(NN + 255) / 256, 256, 0, stream>>>(bat, out + o_batc, out + o_batd);
    topk_kernel<<<NG, EPG, 0, stream>>>(predWs, ei, ea, out,
                                        o_eic, o_eac, o_pwc, o_eid, o_ead, o_pwd);
}
